// ContextEncoderBlock_721554505807
// MI455X (gfx1250) — matrix operand order repaired and checked by reading, not compiled, not run on hardware
//
#include <hip/hip_runtime.h>
#include <hip/hip_bf16.h>

// ---------------- types for CDNA5 WMMA ----------------
typedef __attribute__((ext_vector_type(8)))  __bf16 v8bf;
typedef __attribute__((ext_vector_type(16))) __bf16 v16bf;
typedef __attribute__((ext_vector_type(8)))  float  v8f;

static constexpr int BB   = 8;
static constexpr int TIN  = 2048;
static constexpr int TC   = 2048;
static constexpr int DD   = 1024;
static constexpr int CTXL = 512;
static constexpr float SCALE = 0.03125f; // 1/sqrt(1024)

// ---------------- helpers ----------------
__device__ inline unsigned short f2bf(float f) {
  unsigned int u = __float_as_uint(f);
  unsigned int r = u + 0x7FFFu + ((u >> 16) & 1u); // RNE
  return (unsigned short)(r >> 16);
}
__device__ inline float bf2f(unsigned short h) {
  return __uint_as_float(((unsigned int)h) << 16);
}

// A fragment: 16x32 (MxK) bf16. Lanes 0-15 hold K {0..7,16..23},
// lanes 16-31 hold K {8..15,24..31} of row M = lane%16.
__device__ inline v16bf load_frag_a(const unsigned short* Aptr, int lda) {
  int lane = threadIdx.x & 31;
  const unsigned short* p = Aptr + (size_t)(lane & 15) * lda + (lane >> 4) * 8;
  v8bf lo  = *(const v8bf*)(p);
  v8bf hi8 = *(const v8bf*)(p + 16);
  return __builtin_shufflevector(lo, hi8, 0,1,2,3,4,5,6,7,8,9,10,11,12,13,14,15);
}

// B fragment: 32x16 (KxN) bf16 column-major == rows of weight/key matrix.
// lanes 0-15: K {0..7,16..23} of col n=lane, lanes 16-31: K {8..15,24..31}.
__device__ inline v16bf load_frag_b(const unsigned short* Bptr, int ldb) {
  int lane = threadIdx.x & 31;
  const unsigned short* p = Bptr + (size_t)(lane & 15) * ldb + (lane >> 4) * 8;
  v8bf lo  = *(const v8bf*)(p);
  v8bf hi8 = *(const v8bf*)(p + 16);
  return __builtin_shufflevector(lo, hi8, 0,1,2,3,4,5,6,7,8,9,10,11,12,13,14,15);
}

__device__ inline v8f wmma_bf16(v16bf a, v16bf b, v8f c) {
  return __builtin_amdgcn_wmma_f32_16x16x32_bf16(false, a, false, b,
                                                 (short)0, c, false, false);
}

// ---------------- kernel 1: final = emb_sem[input_ids] + emb_in_pos ----------------
__global__ __launch_bounds__(256) void k_final(const int* __restrict__ input_ids,
                                               const float* __restrict__ emb_sem,
                                               const float* __restrict__ emb_pos,
                                               float* __restrict__ out) {
  size_t idx = (size_t)blockIdx.x * blockDim.x + threadIdx.x; // over B*T*D/4
  int dq = (int)(idx % (DD / 4));
  size_t bt = idx / (DD / 4);
  int t = (int)(bt % TIN);
  size_t b = bt / TIN;
  int tok = input_ids[b * TIN + t];
  const float4 s = *(const float4*)(emb_sem + (size_t)tok * DD + dq * 4);
  const float4 p = *(const float4*)(emb_pos + (size_t)t * DD + dq * 4);
  float4 o; o.x = s.x + p.x; o.y = s.y + p.y; o.z = s.z + p.z; o.w = s.w + p.w;
  *(float4*)(out + bt * DD + dq * 4) = o;
}

// ---------------- kernel 2: fp32 weight -> bf16 ----------------
__global__ __launch_bounds__(256) void k_convw(const float* __restrict__ W,
                                               unsigned short* __restrict__ Wb,
                                               int n) {
  int i = blockIdx.x * 256 + threadIdx.x;
  if (i < n) Wb[i] = f2bf(W[i]);
}

// ---------------- kernel 3: ce (gather + lerped pos) -> bf16, plus vw dot ----------------
__global__ __launch_bounds__(256) void k_ce(const int* __restrict__ ctx_ids,
                                            const float* __restrict__ emb_sem,
                                            const float* __restrict__ emb_cpos,
                                            const float* __restrict__ Vw_w,
                                            const float* __restrict__ Vw_b,
                                            unsigned short* __restrict__ ce_bf,
                                            float* __restrict__ vw_out) {
  __shared__ float red[256];
  int t = blockIdx.x % TC;
  int b = blockIdx.x / TC;
  int tok = ctx_ids[(size_t)b * TC + t];
  float pos_f = (float)t * ((float)(CTXL - 1) / (float)TC);
  int pidx = (int)pos_f;
  float w = pos_f - (float)pidx;
  const float* s0 = emb_cpos + (size_t)pidx * DD;
  const float* s1 = emb_cpos + (size_t)(pidx + 1) * DD;
  const float* se = emb_sem + (size_t)tok * DD;
  unsigned short* cerow = ce_bf + ((size_t)b * TC + t) * DD;
  float part = 0.f;
#pragma unroll
  for (int j = 0; j < 4; ++j) {
    int d = threadIdx.x + j * 256;
    float cv = se[d] + (s0[d] * (1.f - w) + s1[d] * w);
    cerow[d] = f2bf(cv);
    part += cv * Vw_w[d];
  }
  red[threadIdx.x] = part;
  __syncthreads();
  for (int s = 128; s > 0; s >>= 1) {
    if ((int)threadIdx.x < s) red[threadIdx.x] += red[threadIdx.x + s];
    __syncthreads();
  }
  if (threadIdx.x == 0) vw_out[(size_t)b * TC + t] = red[0] + Vw_b[0];
}

// ---------------- kernel 4: O[m,n] = sum_d A[m,d] * W[n,d]  (bf16 WMMA) ----------------
// 256 threads = 8 waves; each wave owns a 32x64 output strip (2x4 wmma tiles).
// All fragments loaded into distinct regs first so the scheduler can issue one
// load clause and use partial s_wait_loadcnt before each wmma.
__global__ __launch_bounds__(256) void k_gemm(const unsigned short* __restrict__ A,
                                              const unsigned short* __restrict__ W,
                                              unsigned short* __restrict__ O,
                                              int M) {
  const int ntiles = DD / 64; // 16
  int mt = blockIdx.x / ntiles;
  int nt = blockIdx.x % ntiles;
  int wave = threadIdx.x >> 5;
  int row0 = mt * 256 + wave * 32; // two 16-row tiles per wave
  int col0 = nt * 64;
  v8f acc[2][4] = {};
  for (int kk = 0; kk < DD; kk += 32) {
    v16bf a0 = load_frag_a(A + (size_t)row0 * DD + kk, DD);
    v16bf a1 = load_frag_a(A + (size_t)(row0 + 16) * DD + kk, DD);
    v16bf b0 = load_frag_b(W + (size_t)(col0 +  0) * DD + kk, DD);
    v16bf b1 = load_frag_b(W + (size_t)(col0 + 16) * DD + kk, DD);
    v16bf b2 = load_frag_b(W + (size_t)(col0 + 32) * DD + kk, DD);
    v16bf b3 = load_frag_b(W + (size_t)(col0 + 48) * DD + kk, DD);
    __builtin_prefetch(A + (size_t)row0 * DD + kk + 32, 0, 1);
    acc[0][0] = wmma_bf16(a0, b0, acc[0][0]);
    acc[0][1] = wmma_bf16(a0, b1, acc[0][1]);
    acc[0][2] = wmma_bf16(a0, b2, acc[0][2]);
    acc[0][3] = wmma_bf16(a0, b3, acc[0][3]);
    acc[1][0] = wmma_bf16(a1, b0, acc[1][0]);
    acc[1][1] = wmma_bf16(a1, b1, acc[1][1]);
    acc[1][2] = wmma_bf16(a1, b2, acc[1][2]);
    acc[1][3] = wmma_bf16(a1, b3, acc[1][3]);
  }
  int lane = threadIdx.x & 31;
  int n = lane & 15, hi = lane >> 4;
#pragma unroll
  for (int u = 0; u < 2; ++u)
#pragma unroll
    for (int t = 0; t < 4; ++t)
#pragma unroll
      for (int i = 0; i < 8; ++i) {
        int r = row0 + u * 16 + hi * 8 + i;
        O[(size_t)r * DD + col0 + t * 16 + n] = f2bf(acc[u][t][i]);
      }
}

// ---------------- kernel 5: attention pass 1 — softmax stats + attn-weight dot ----------------
// Block = (b, 128-row q tile). Online softmax over k tiles of 64.
// Outputs m[q] (row max of S*scale) and g[q] = aw[q]/l[q] = aw_r/l_r^2.
__global__ __launch_bounds__(256) void k_attn1(const unsigned short* __restrict__ Q,
                                               const unsigned short* __restrict__ K,
                                               const float* __restrict__ vw,
                                               float* __restrict__ m_out,
                                               float* __restrict__ g_out) {
  __shared__ float S[128][65];
  int qt = blockIdx.x % (TC / 128);
  int b  = blockIdx.x / (TC / 128);
  const unsigned short* Qb = Q + (size_t)b * TC * DD;
  const unsigned short* Kb = K + (size_t)b * TC * DD;
  int wave = threadIdx.x >> 5;
  int lane = threadIdx.x & 31;
  int srow = wave * 16;
  int q0 = qt * 128;
  float m_r = -1e30f, l_r = 0.f, aw_r = 0.f;

  for (int kt = 0; kt < TC; kt += 64) {
    v8f acc[4] = {};
    for (int kk = 0; kk < DD; kk += 32) {
      v16bf a  = load_frag_a(Qb + (size_t)(q0 + srow) * DD + kk, DD);
      v16bf b0 = load_frag_b(Kb + (size_t)(kt +  0) * DD + kk, DD);
      v16bf b1 = load_frag_b(Kb + (size_t)(kt + 16) * DD + kk, DD);
      v16bf b2 = load_frag_b(Kb + (size_t)(kt + 32) * DD + kk, DD);
      v16bf b3 = load_frag_b(Kb + (size_t)(kt + 48) * DD + kk, DD);
      acc[0] = wmma_bf16(a, b0, acc[0]);
      acc[1] = wmma_bf16(a, b1, acc[1]);
      acc[2] = wmma_bf16(a, b2, acc[2]);
      acc[3] = wmma_bf16(a, b3, acc[3]);
    }
    int n = lane & 15, hi = lane >> 4;
#pragma unroll
    for (int t = 0; t < 4; ++t)
#pragma unroll
      for (int i = 0; i < 8; ++i)
        S[srow + hi * 8 + i][t * 16 + n] = acc[t][i];
    __syncthreads();
    if (threadIdx.x < 128) {
      int r = threadIdx.x;
      float tmax = -1e30f;
      for (int c = 0; c < 64; ++c) tmax = fmaxf(tmax, S[r][c] * SCALE);
      float mn = fmaxf(m_r, tmax);
      float corr = __expf(m_r - mn);
      l_r *= corr; aw_r *= corr;
      for (int c = 0; c < 64; ++c) {
        float p = __expf(S[r][c] * SCALE - mn);
        l_r += p;
        aw_r += p * vw[(size_t)b * TC + kt + c];
      }
      m_r = mn;
    }
    __syncthreads();
  }
  if (threadIdx.x < 128) {
    size_t qi = (size_t)b * TC + q0 + threadIdx.x;
    m_out[qi] = m_r;
    g_out[qi] = aw_r / (l_r * l_r);
  }
}

// ---------------- kernel 6: attention pass 2 — c[b,k] = sum_q g[q]*exp(S-m[q]) ----------------
// Block = (b, 128-col k tile). Loop q tiles of 64. S tile 64x128 in LDS.
__global__ __launch_bounds__(256) void k_attn2(const unsigned short* __restrict__ Q,
                                               const unsigned short* __restrict__ K,
                                               const float* __restrict__ m_in,
                                               const float* __restrict__ g_in,
                                               float* __restrict__ c_out) {
  __shared__ float S[64][129];
  __shared__ float gs[64], ms[64];
  int ktile = blockIdx.x % (TC / 128);
  int b = blockIdx.x / (TC / 128);
  int k0 = ktile * 128;
  const unsigned short* Qb = Q + (size_t)b * TC * DD;
  const unsigned short* Kb = K + (size_t)b * TC * DD;
  int wave = threadIdx.x >> 5;
  int lane = threadIdx.x & 31;
  int srow = (wave & 3) * 16;   // q strip within 64
  int scol = (wave >> 2) * 64;  // column half of 128
  float c_acc = 0.f;

  for (int q0 = 0; q0 < TC; q0 += 64) {
    v8f acc[4] = {};
    for (int kk = 0; kk < DD; kk += 32) {
      v16bf a  = load_frag_a(Qb + (size_t)(q0 + srow) * DD + kk, DD);
      v16bf b0 = load_frag_b(Kb + (size_t)(k0 + scol +  0) * DD + kk, DD);
      v16bf b1 = load_frag_b(Kb + (size_t)(k0 + scol + 16) * DD + kk, DD);
      v16bf b2 = load_frag_b(Kb + (size_t)(k0 + scol + 32) * DD + kk, DD);
      v16bf b3 = load_frag_b(Kb + (size_t)(k0 + scol + 48) * DD + kk, DD);
      acc[0] = wmma_bf16(a, b0, acc[0]);
      acc[1] = wmma_bf16(a, b1, acc[1]);
      acc[2] = wmma_bf16(a, b2, acc[2]);
      acc[3] = wmma_bf16(a, b3, acc[3]);
    }
    int n = lane & 15, hi = lane >> 4;
#pragma unroll
    for (int t = 0; t < 4; ++t)
#pragma unroll
      for (int i = 0; i < 8; ++i)
        S[srow + hi * 8 + i][scol + t * 16 + n] = acc[t][i];
    if (threadIdx.x < 64) {
      size_t qi = (size_t)b * TC + q0 + threadIdx.x;
      gs[threadIdx.x] = g_in[qi];
      ms[threadIdx.x] = m_in[qi];
    }
    __syncthreads();
    if (threadIdx.x < 128) {
      int kc = threadIdx.x;
      for (int q = 0; q < 64; ++q)
        c_acc += gs[q] * __expf(S[q][kc] * SCALE - ms[q]);
    }
    __syncthreads();
  }
  if (threadIdx.x < 128)
    c_out[(size_t)b * TC + k0 + threadIdx.x] = c_acc;
}

// ---------------- kernel 7: agg[b,d] = sum_k c[b,k] * v[b,k,d] ----------------
__global__ __launch_bounds__(256) void k_agg(const float* __restrict__ c_in,
                                             const unsigned short* __restrict__ V,
                                             float* __restrict__ agg) {
  int b = blockIdx.x / (DD / 256);
  int d = (blockIdx.x % (DD / 256)) * 256 + threadIdx.x;
  const unsigned short* Vb = V + (size_t)b * TC * DD;
  const float* cb = c_in + (size_t)b * TC;
  float s = 0.f;
  for (int k = 0; k < TC; ++k)
    s += cb[k] * bf2f(Vb[(size_t)k * DD + d]);
  agg[b * DD + d] = s;
}

// ---------------- kernel 8: out[b,0,:] += agg @ Wt^T + bt ----------------
__global__ __launch_bounds__(256) void k_trans(const float* __restrict__ agg,
                                               const float* __restrict__ Wt,
                                               const float* __restrict__ bt,
                                               float* __restrict__ out) {
  int idx = blockIdx.x * 256 + threadIdx.x; // b*DD + e
  int b = idx / DD, e = idx % DD;
  float s = bt[e];
  const float* ag = agg + (size_t)b * DD;
  const float* wr = Wt + (size_t)e * DD;
  for (int d = 0; d < DD; ++d) s += ag[d] * wr[d];
  out[(size_t)b * TIN * DD + e] += s;
}

// ---------------- launch ----------------
extern "C" void kernel_launch(void* const* d_in, const int* in_sizes, int n_in,
                              void* d_out, int out_size, void* d_ws, size_t ws_size,
                              hipStream_t stream) {
  const int*   ctx_ids   = (const int*)d_in[0];
  const int*   input_ids = (const int*)d_in[1];
  const float* emb_sem   = (const float*)d_in[2];
  const float* emb_ipos  = (const float*)d_in[3];
  const float* emb_cpos  = (const float*)d_in[4];
  const float* Wq        = (const float*)d_in[5];
  const float* Wk        = (const float*)d_in[6];
  const float* Wv        = (const float*)d_in[7];
  const float* Vw_w      = (const float*)d_in[8];
  const float* Vw_b      = (const float*)d_in[9];
  const float* Wt        = (const float*)d_in[10];
  const float* bt        = (const float*)d_in[11];
  float* out = (float*)d_out;

  // workspace layout
  char* w = (char*)d_ws;
  const size_t SZ_MAT  = (size_t)BB * TC * DD * sizeof(unsigned short); // 32 MiB
  const size_t SZ_W    = (size_t)DD * DD * sizeof(unsigned short);      // 2 MiB
  const size_t SZ_ROW  = (size_t)BB * TC * sizeof(float);               // 64 KiB
  unsigned short* ce_bf = (unsigned short*)(w);                 size_t off = SZ_MAT;
  unsigned short* q_bf  = (unsigned short*)(w + off);           off += SZ_MAT;
  unsigned short* k_bf  = (unsigned short*)(w + off);           off += SZ_MAT;
  unsigned short* v_bf  = (unsigned short*)(w + off);           off += SZ_MAT;
  unsigned short* wq_bf = (unsigned short*)(w + off);           off += SZ_W;
  unsigned short* wk_bf = (unsigned short*)(w + off);           off += SZ_W;
  unsigned short* wv_bf = (unsigned short*)(w + off);           off += SZ_W;
  float* vw   = (float*)(w + off);                              off += SZ_ROW;
  float* m_ws = (float*)(w + off);                              off += SZ_ROW;
  float* g_ws = (float*)(w + off);                              off += SZ_ROW;
  float* c_ws = (float*)(w + off);                              off += SZ_ROW;
  float* agg  = (float*)(w + off);                              off += (size_t)BB * DD * sizeof(float);
  (void)ws_size; (void)in_sizes; (void)n_in; (void)out_size;

  // 1) input branch -> d_out
  k_final<<<(BB * TIN * DD / 4) / 256, 256, 0, stream>>>(input_ids, emb_sem, emb_ipos, out);

  // 2) weights -> bf16
  k_convw<<<(DD * DD + 255) / 256, 256, 0, stream>>>(Wq, wq_bf, DD * DD);
  k_convw<<<(DD * DD + 255) / 256, 256, 0, stream>>>(Wk, wk_bf, DD * DD);
  k_convw<<<(DD * DD + 255) / 256, 256, 0, stream>>>(Wv, wv_bf, DD * DD);

  // 3) ce (bf16) + vw
  k_ce<<<BB * TC, 256, 0, stream>>>(ctx_ids, emb_sem, emb_cpos, Vw_w, Vw_b, ce_bf, vw);

  // 4) q, k, v GEMMs via WMMA (wave = 32x64 strip)
  const int M = BB * TC;
  const int gemm_blocks = (M / 256) * (DD / 64);
  k_gemm<<<gemm_blocks, 256, 0, stream>>>(ce_bf, wq_bf, q_bf, M);
  k_gemm<<<gemm_blocks, 256, 0, stream>>>(ce_bf, wk_bf, k_bf, M);
  k_gemm<<<gemm_blocks, 256, 0, stream>>>(ce_bf, wv_bf, v_bf, M);

  // 5) attention pass 1: softmax stats + attn-weight scalar per row
  k_attn1<<<BB * (TC / 128), 256, 0, stream>>>(q_bf, k_bf, vw, m_ws, g_ws);

  // 6) attention pass 2: c[b,k]
  k_attn2<<<BB * (TC / 128), 256, 0, stream>>>(q_bf, k_bf, m_ws, g_ws, c_ws);

  // 7) agg = c @ v
  k_agg<<<BB * (DD / 256), 256, 0, stream>>>(c_ws, v_bf, agg);

  // 8) out[:,0,:] += agg @ Wt^T + bt
  k_trans<<<(BB * DD) / 256, 256, 0, stream>>>(agg, Wt, bt, out);
}
